// CorrelationModule_25692494365415
// MI455X (gfx1250) — compile-verified, matrix operand order checked by reading, not run on hardware
//
#include <hip/hip_runtime.h>
#include <cstdint>
#include <cstddef>

// ---------------------------------------------------------------------------
// MI455X (gfx1250) correlation module.
// Compute-bound (~73 GFLOP vs ~4MB inputs) -> route the big convs through
// v_wmma_f32_16x16x32_bf16 as implicit GEMMs (wave32, 16x16 tiles), and feed
// the LDS B-tiles with the CDNA5 direct global->LDS async path
// (global_load_async_to_lds_b128 / ASYNCcnt), double-buffered.
// K dimension padded to multiples of KCB so the stage body is a fully
// unrolled 4x WMMA sequence (zero A-fragments/LDS for pad chunks).
// ---------------------------------------------------------------------------

typedef __bf16 bf16;
typedef __attribute__((ext_vector_type(16))) __bf16 v16bf;
typedef __attribute__((ext_vector_type(8)))  float  v8f;

#define EPSV 1e-5f

static constexpr int P   = 49;    // D*D pairs
static constexpr int H1  = 40, W1 = 56, HW1 = H1 * W1;   // 2240
static constexpr int H2  = 20, W2 = 28, HW2 = H2 * W2;   // 560
static constexpr int C0  = 258;   // 2*FEAT + 2
static constexpr int KC0 = 9;     // chunks present in stacked input X0
static constexpr int KCB = 4;     // K chunks per pipeline stage (unrolled)

__device__ __forceinline__ float bf2f(bf16 x) { return (float)x; }
__device__ __forceinline__ bf16  f2bf(float x) { return (bf16)x; }

// ---------------------------------------------------------------------------
// Weight prep: fold BN (scale = g/sqrt(v+eps)) into weights, reorder OIHW f32
// weights into per-(tap, kchunk, mtile) WMMA A-fragments:
//   frag[(t*KCP + kc)*MT + mt][lane][i] , lane: M = lane&15, Khalf = lane>>4
// KCP is the KCB-padded chunk count; chunks with k >= Cin become zeros.
// ---------------------------------------------------------------------------
__global__ void prep_wfrag(const float* __restrict__ w,
                           const float* __restrict__ g, const float* __restrict__ va,
                           int Cout, int Cin, int taps, int KCP, int MT,
                           bf16* __restrict__ WF)
{
    int total = taps * KCP * MT * 512;
    for (int idx = blockIdx.x * blockDim.x + threadIdx.x; idx < total;
         idx += gridDim.x * blockDim.x) {
        int i    = idx & 15;
        int lane = (idx >> 4) & 31;
        int rest = idx >> 9;
        int mt = rest % MT; rest /= MT;
        int kc = rest % KCP;
        int t  = rest / KCP;
        int m = mt * 16 + (lane & 15);
        int k = kc * 32 + (lane >> 4) * 16 + i;
        float val = 0.f;
        if (m < Cout && k < Cin) {
            float s = 1.f;
            if (g) s = g[m] * rsqrtf(va[m] + EPSV);
            val = w[((size_t)m * Cin + k) * taps + t] * s;
        }
        WF[idx] = f2bf(val);
    }
}

__global__ void prep_bias(const float* __restrict__ g, const float* __restrict__ be,
                          const float* __restrict__ mu, const float* __restrict__ va,
                          const float* __restrict__ pb, int Cout, int Mpad,
                          float* __restrict__ out)
{
    int m = blockIdx.x * blockDim.x + threadIdx.x;
    if (m >= Mpad) return;
    float b = 0.f;
    if (m < Cout) {
        if (g) { float s = g[m] * rsqrtf(va[m] + EPSV); b = be[m] - mu[m] * s; }
        else   { b = pb[m]; }
    }
    out[m] = b;
}

// ---------------------------------------------------------------------------
// Build stacked input X0: layout [p][kc(9)][pix(2240)][32ch] bf16.
//   ch 0..127   : f1
//   ch 128..255 : bilinear sample of f2 at coords + (dx,dy)
//   ch 256/257  : dx, dy ; ch 258..287 : zero pad
// ---------------------------------------------------------------------------
__global__ void build_stack(const float* __restrict__ f1, const float* __restrict__ f2,
                            const float* __restrict__ coords, bf16* __restrict__ X0,
                            int total)
{
    int u = blockIdx.x * blockDim.x + threadIdx.x;
    if (u >= total) return;
    int qq  = u % 36;
    int r   = u / 36;
    int pix = r % HW1;
    int p   = r / HW1;
    int cbase = qq * 8;
    float offx = (float)(p / 7) - 3.0f;
    float offy = (float)(p % 7) - 3.0f;

    float vals[8];
    if (cbase + 8 <= 128) {
        #pragma unroll
        for (int i = 0; i < 8; ++i) vals[i] = f1[(size_t)(cbase + i) * HW1 + pix];
    } else if (cbase >= 128 && cbase + 8 <= 256) {
        float sx = coords[pix] + offx;
        float sy = coords[HW1 + pix] + offy;
        float x0f = floorf(sx), y0f = floorf(sy);
        int x0 = (int)x0f, y0 = (int)y0f, x1 = x0 + 1, y1 = y0 + 1;
        float wx = sx - x0f, wy = sy - y0f;
        bool vx0 = (x0 >= 0) & (x0 < W1), vx1 = (x1 >= 0) & (x1 < W1);
        bool vy0 = (y0 >= 0) & (y0 < H1), vy1 = (y1 >= 0) & (y1 < H1);
        int cx0 = min(max(x0, 0), W1 - 1), cx1 = min(max(x1, 0), W1 - 1);
        int cy0 = min(max(y0, 0), H1 - 1), cy1 = min(max(y1, 0), H1 - 1);
        int i00 = cy0 * W1 + cx0, i10 = cy0 * W1 + cx1;
        int i01 = cy1 * W1 + cx0, i11 = cy1 * W1 + cx1;
        float w00 = (1.f - wx) * (1.f - wy) * (vx0 & vy0 ? 1.f : 0.f);
        float w10 = wx * (1.f - wy) * (vx1 & vy0 ? 1.f : 0.f);
        float w01 = (1.f - wx) * wy * (vx0 & vy1 ? 1.f : 0.f);
        float w11 = wx * wy * (vx1 & vy1 ? 1.f : 0.f);
        #pragma unroll
        for (int i = 0; i < 8; ++i) {
            size_t cb = (size_t)(cbase - 128 + i) * HW1;
            vals[i] = w00 * f2[cb + i00] + w10 * f2[cb + i10] +
                      w01 * f2[cb + i01] + w11 * f2[cb + i11];
        }
    } else {
        #pragma unroll
        for (int i = 0; i < 8; ++i) {
            int c = cbase + i;
            vals[i] = (c == 256) ? offx : (c == 257) ? offy : 0.f;
        }
    }

    __attribute__((aligned(16))) bf16 out8[8];
    #pragma unroll
    for (int i = 0; i < 8; ++i) out8[i] = f2bf(vals[i]);
    int kc = qq >> 2, quarter = qq & 3;
    bf16* dst = X0 + ((size_t)(p * KC0 + kc) * HW1 + pix) * 32 + quarter * 8;
    *(uint4*)dst = *(const uint4*)out8;
}

// ---------------------------------------------------------------------------
// Implicit-GEMM conv via WMMA bf16 (3x3 convs, strided conv, 1x1 convs).
// Block: 256 thr = 8 waves; wave grid 2(M) x 4(N) -> 32 out-ch x 64 pixels.
// Grid: (Mpad/32, Hout, P). Pixel slot n covers one output row (Wout<=64).
// Pipeline: stages = taps x (KCP/KCB); per stage the next B-tile is pulled
// directly into the other LDS buffer with global_load_async_to_lds_b128
// (ASYNCcnt) while the current buffer feeds exactly KCB unrolled WMMAs.
// KCx = chunks actually present in X (pad chunks -> LDS zero-fill).
// ---------------------------------------------------------------------------
__global__ __launch_bounds__(256)
void conv_gemm_wmma(const bf16* __restrict__ X, int xPstride, int Hin, int Win,
                    const bf16* __restrict__ WF, const float* __restrict__ bias,
                    bf16* __restrict__ Y, int yPstride, int Hout, int Wout,
                    int KCx, int KCP, int MT, int taps, int stride, int pad,
                    int relu)
{
    __shared__ __attribute__((aligned(32))) bf16 lds[2][64 * KCB * 32];  // 32 KB

    const int p    = blockIdx.z;
    const int y0   = blockIdx.y;
    const int tid  = threadIdx.x;
    const int lane = tid & 31;
    const int wave = tid >> 5;
    const int mw   = wave >> 2;       // 0..1
    const int nw   = wave & 3;        // 0..3
    const int mt   = blockIdx.x * 2 + mw;

    const bf16* Xp = X + (size_t)p * xPstride;
    const int nsk     = KCP / KCB;    // KCP is a multiple of KCB
    const int nstages = taps * nsk;
    v8f acc = {};

    // issue one stage's B-tile: async global->LDS for valid units, ds zero-fill
    // for OOB / K-pad units. Tile layout: [pix(64)][kc_local(KCB)][32ch] bf16.
    auto issue_stage = [&](int s, int buf) {
        int t  = s / nsk;
        int kb = (s - t * nsk) * KCB;
        int ky = (taps == 9) ? (t / 3) : 0;
        int kx = (taps == 9) ? (t % 3) : 0;
        int iy = y0 * stride + ky - pad;
        #pragma unroll
        for (int u = tid; u < 64 * KCB * 4; u += 256) {
            int pix     = u >> 4;          // 16 units per pixel
            int q       = u & 15;
            int kcl     = q >> 2;
            int quarter = q & 3;
            int ix = pix * stride + kx - pad;
            bf16* dst = &lds[buf][(pix * KCB + kcl) * 32 + quarter * 8];
            if (pix < Wout && (kb + kcl) < KCx &&
                ix >= 0 && ix < Win && iy >= 0 && iy < Hin) {
                const bf16* src = Xp + ((size_t)(kb + kcl) * Hin * Win
                                        + iy * Win + ix) * 32 + quarter * 8;
                unsigned           laddr = (unsigned)(uintptr_t)dst;  // LDS byte addr
                unsigned long long gaddr = (unsigned long long)(uintptr_t)src;
                asm volatile("global_load_async_to_lds_b128 %0, %1, off"
                             :: "v"(laddr), "v"(gaddr) : "memory");
            } else {
                *(uint4*)dst = make_uint4(0u, 0u, 0u, 0u);
            }
        }
    };

    issue_stage(0, 0);

    for (int s = 0; s < nstages; ++s) {
        const int buf = s & 1;
        asm volatile("s_wait_asynccnt 0x0" ::: "memory");  // own tile landed
        __syncthreads();                                   // whole tile visible
        if (s + 1 < nstages) issue_stage(s + 1, buf ^ 1);  // overlap next fill

        int t  = s / nsk;
        int kb = (s - t * nsk) * KCB;
        const bf16* wf = WF + ((size_t)(t * KCP + kb)) * MT * 512;
        __builtin_prefetch(wf + (size_t)KCB * MT * 512, 0, 1);
        #pragma unroll
        for (int kc = 0; kc < KCB; ++kc) {
            v16bf a = *(const v16bf*)(wf + ((size_t)kc * MT + mt) * 512 + lane * 16);
            v16bf b = *(const v16bf*)(&lds[buf][((nw * 16 + (lane & 15)) * KCB + kc) * 32
                                               + (lane >> 4) * 16]);
            acc = __builtin_amdgcn_wmma_f32_16x16x32_bf16(
                false, a, false, b, (short)0, acc, false, false);
        }
    }

    // --- epilogue: bias + ReLU + bf16 store (lane holds 8 consecutive M) ---
    const int g0  = mt * 16 + (lane >> 4) * 8;
    const int pix = nw * 16 + (lane & 15);
    if (pix < Wout) {
        __attribute__((aligned(16))) bf16 out8[8];
        #pragma unroll
        for (int v = 0; v < 8; ++v) {
            float f = acc[v] + bias[g0 + v];
            if (relu) f = fmaxf(f, 0.f);
            out8[v] = f2bf(f);
        }
        int chunk = g0 >> 5;
        bf16* dst = Y + (size_t)p * yPstride
                      + ((size_t)chunk * Hout * Wout + y0 * Wout + pix) * 32
                      + (g0 & 31);
        *(uint4*)dst = *(const uint4*)out8;
    }
}

// ---------------------------------------------------------------------------
// Transposed conv 4x4 / lhs_dilation 2 / pad 2 : 64 -> 32 ch (parity taps).
// ---------------------------------------------------------------------------
__global__ void deconv_block_k(const bf16* __restrict__ X4, const float* __restrict__ w5,
                               const float* __restrict__ g5, const float* __restrict__ b5,
                               const float* __restrict__ m5, const float* __restrict__ v5,
                               bf16* __restrict__ X5)
{
    int u = blockIdx.x * blockDim.x + threadIdx.x;
    if (u >= P * HW1 * 32) return;
    int oc  = u & 31;
    int pix = (u >> 5) % HW1;
    int p   = (u >> 5) / HW1;
    int y = pix / W1, x = pix % W1;

    float acc = 0.f;
    for (int ky = 0; ky < 4; ++ky) {
        int jy = y + ky - 2;
        if (jy < 0 || (jy & 1)) continue;
        int iy = jy >> 1;
        if (iy >= H2) continue;
        for (int kx = 0; kx < 4; ++kx) {
            int jx = x + kx - 2;
            if (jx < 0 || (jx & 1)) continue;
            int ix = jx >> 1;
            if (ix >= W2) continue;
            int ipix = iy * W2 + ix;
            for (int ic = 0; ic < 64; ++ic) {
                float xv = bf2f(X4[((size_t)(p * 2 + (ic >> 5)) * HW2 + ipix) * 32
                                   + (ic & 31)]);
                acc += xv * w5[(((size_t)oc * 64 + ic) * 4 + ky) * 4 + kx];
            }
        }
    }
    float s = g5[oc] * rsqrtf(v5[oc] + EPSV);
    float o = fmaxf(acc * s + (b5[oc] - m5[oc] * s), 0.f);
    X5[((size_t)p * HW1 + pix) * 32 + oc] = f2bf(o);
}

// ---------------------------------------------------------------------------
// Final 3x3 conv 32 -> 1 (cost); writes cost to ws and to output ch 0..48.
// ---------------------------------------------------------------------------
__global__ void conv6_cost(const bf16* __restrict__ X5, const float* __restrict__ w6,
                           const float* __restrict__ b6, float* __restrict__ costws,
                           float* __restrict__ out)
{
    int u = blockIdx.x * blockDim.x + threadIdx.x;
    if (u >= P * HW1) return;
    int pix = u % HW1, p = u / HW1;
    int y = pix / W1, x = pix % W1;
    float acc = b6[0];
    for (int t = 0; t < 9; ++t) {
        int iy = y + t / 3 - 1, ix = x + t % 3 - 1;
        if (iy < 0 || iy >= H1 || ix < 0 || ix >= W1) continue;
        int ipix = iy * W1 + ix;
        for (int c = 0; c < 32; ++c)
            acc += bf2f(X5[((size_t)p * HW1 + ipix) * 32 + c]) * w6[c * 9 + t];
    }
    costws[(size_t)p * HW1 + pix] = acc;
    out[(size_t)p * HW1 + pix]    = acc;   // cost channels of output
}

// ---------------------------------------------------------------------------
// Per-pixel DAP (49x49) -> softmax -> embedding reduction (32 ch).
// ---------------------------------------------------------------------------
__global__ void dap_softmax_emb(const float* __restrict__ dap,
                                const float* __restrict__ cost,
                                const bf16* __restrict__ E3,
                                float* __restrict__ out)
{
    int pix = blockIdx.x * blockDim.x + threadIdx.x;
    if (pix >= HW1) return;
    float c[P], s[P];
    for (int q = 0; q < P; ++q) c[q] = cost[(size_t)q * HW1 + pix];
    float mx = -3.4e38f;
    for (int p = 0; p < P; ++p) {
        float a = 0.f;
        for (int q = 0; q < P; ++q) a += dap[p * P + q] * c[q];
        s[p] = a;
        mx = fmaxf(mx, a);
    }
    float sum = 0.f;
    for (int p = 0; p < P; ++p) { s[p] = __expf(s[p] - mx); sum += s[p]; }
    float inv = 1.f / sum;
    for (int e = 0; e < 32; ++e) {
        float a = 0.f;
        for (int p = 0; p < P; ++p)
            a += s[p] * bf2f(E3[((size_t)p * HW1 + pix) * 32 + e]);
        out[(size_t)(P + e) * HW1 + pix] = a * inv;
    }
}

// ---------------------------------------------------------------------------
// Host launcher. Input order (setup_inputs insertion order; params flattened
// as a JAX pytree, dict keys sorted):
//  0:f1 1:f2 2:coords 3:dap
//  4:emb.b1 5:emb.b2 6:emb.b3 7:emb.w1 8:emb.w2 9:emb.w3
//  10:mnet.b6 11-14:bn1(g,b,m,v) 15-18:bn2 19-22:bn3 23-26:bn4 27-30:bn5
//  31:mnet.w1 32:w2 33:w3 34:w4 35:w5 36:w6
// ---------------------------------------------------------------------------
extern "C" void kernel_launch(void* const* d_in, const int* in_sizes, int n_in,
                              void* d_out, int out_size, void* d_ws, size_t ws_size,
                              hipStream_t stream)
{
    const float* f1     = (const float*)d_in[0];
    const float* f2     = (const float*)d_in[1];
    const float* coords = (const float*)d_in[2];
    const float* dap    = (const float*)d_in[3];
    const float* e_b1 = (const float*)d_in[4];
    const float* e_b2 = (const float*)d_in[5];
    const float* e_b3 = (const float*)d_in[6];
    const float* e_w1 = (const float*)d_in[7];
    const float* e_w2 = (const float*)d_in[8];
    const float* e_w3 = (const float*)d_in[9];
    const float* b6   = (const float*)d_in[10];
    const float* bn1g = (const float*)d_in[11]; const float* bn1b = (const float*)d_in[12];
    const float* bn1m = (const float*)d_in[13]; const float* bn1v = (const float*)d_in[14];
    const float* bn2g = (const float*)d_in[15]; const float* bn2b = (const float*)d_in[16];
    const float* bn2m = (const float*)d_in[17]; const float* bn2v = (const float*)d_in[18];
    const float* bn3g = (const float*)d_in[19]; const float* bn3b = (const float*)d_in[20];
    const float* bn3m = (const float*)d_in[21]; const float* bn3v = (const float*)d_in[22];
    const float* bn4g = (const float*)d_in[23]; const float* bn4b = (const float*)d_in[24];
    const float* bn4m = (const float*)d_in[25]; const float* bn4v = (const float*)d_in[26];
    const float* bn5g = (const float*)d_in[27]; const float* bn5b = (const float*)d_in[28];
    const float* bn5m = (const float*)d_in[29]; const float* bn5v = (const float*)d_in[30];
    const float* w1 = (const float*)d_in[31];
    const float* w2 = (const float*)d_in[32];
    const float* w3 = (const float*)d_in[33];
    const float* w4 = (const float*)d_in[34];
    const float* w5 = (const float*)d_in[35];
    const float* w6 = (const float*)d_in[36];
    float* out = (float*)d_out;

    // bump allocator over workspace
    char*  ws  = (char*)d_ws;
    size_t off = 0;
    auto alloc = [&](size_t bytes) -> char* {
        char* r = ws + off;
        off = (off + bytes + 255) & ~(size_t)255;
        return r;
    };

    bf16* X0 = (bf16*)alloc((size_t)P * 9 * HW1 * 32 * 2);
    bf16* Y1 = (bf16*)alloc((size_t)P * 3 * HW1 * 32 * 2);
    bf16* Y2 = (bf16*)alloc((size_t)P * 4 * HW2 * 32 * 2);
    bf16* Y3 = (bf16*)alloc((size_t)P * 4 * HW2 * 32 * 2);
    bf16* Y4 = (bf16*)alloc((size_t)P * 2 * HW2 * 32 * 2);
    bf16* X5 = (bf16*)alloc((size_t)P * 1 * HW1 * 32 * 2);
    bf16* E1 = (bf16*)alloc((size_t)P * 2 * HW1 * 32 * 2);
    bf16* E2 = (bf16*)alloc((size_t)P * 2 * HW1 * 32 * 2);
    bf16* E3 = (bf16*)alloc((size_t)P * 1 * HW1 * 32 * 2);
    float* costws = (float*)alloc((size_t)P * HW1 * 4);
    // K-padded fragment buffers (KCP multiple of KCB=4)
    bf16* WF1 = (bf16*)alloc((size_t)9 * 12 * 6 * 512 * 2);
    bf16* WF2 = (bf16*)alloc((size_t)9 *  4 * 8 * 512 * 2);
    bf16* WF3 = (bf16*)alloc((size_t)9 *  4 * 8 * 512 * 2);
    bf16* WF4 = (bf16*)alloc((size_t)9 *  4 * 4 * 512 * 2);
    bf16* WE1 = (bf16*)alloc((size_t)1 * 12 * 4 * 512 * 2);
    bf16* WE2 = (bf16*)alloc((size_t)1 *  4 * 4 * 512 * 2);
    bf16* WE3 = (bf16*)alloc((size_t)1 *  4 * 2 * 512 * 2);
    float* B1  = (float*)alloc(96 * 4);
    float* B2  = (float*)alloc(128 * 4);
    float* B3  = (float*)alloc(128 * 4);
    float* B4  = (float*)alloc(64 * 4);
    float* BE1 = (float*)alloc(64 * 4);
    float* BE2 = (float*)alloc(64 * 4);
    float* BE3 = (float*)alloc(32 * 4);

    auto cdiv = [](int a, int b) { return (a + b - 1) / b; };

    // ---- weight prep (KCP = padded chunk count) ----
    prep_wfrag<<<cdiv(9 * 12 * 6 * 512, 256), 256, 0, stream>>>(w1, bn1g, bn1v, 96, C0, 9, 12, 6, WF1);
    prep_wfrag<<<cdiv(9 *  4 * 8 * 512, 256), 256, 0, stream>>>(w2, bn2g, bn2v, 128, 96, 9, 4, 8, WF2);
    prep_wfrag<<<cdiv(9 *  4 * 8 * 512, 256), 256, 0, stream>>>(w3, bn3g, bn3v, 128, 128, 9, 4, 8, WF3);
    prep_wfrag<<<cdiv(9 *  4 * 4 * 512, 256), 256, 0, stream>>>(w4, bn4g, bn4v, 64, 128, 9, 4, 4, WF4);
    prep_wfrag<<<cdiv(1 * 12 * 4 * 512, 256), 256, 0, stream>>>(e_w1, nullptr, nullptr, 48, C0, 1, 12, 4, WE1);
    prep_wfrag<<<cdiv(1 *  4 * 4 * 512, 256), 256, 0, stream>>>(e_w2, nullptr, nullptr, 64, 48, 1, 4, 4, WE2);
    prep_wfrag<<<cdiv(1 *  4 * 2 * 512, 256), 256, 0, stream>>>(e_w3, nullptr, nullptr, 32, 64, 1, 4, 2, WE3);
    prep_bias<<<2, 64, 0, stream>>>(bn1g, bn1b, bn1m, bn1v, nullptr, 96, 96, B1);
    prep_bias<<<2, 64, 0, stream>>>(bn2g, bn2b, bn2m, bn2v, nullptr, 128, 128, B2);
    prep_bias<<<2, 64, 0, stream>>>(bn3g, bn3b, bn3m, bn3v, nullptr, 128, 128, B3);
    prep_bias<<<1, 64, 0, stream>>>(bn4g, bn4b, bn4m, bn4v, nullptr, 64, 64, B4);
    prep_bias<<<1, 64, 0, stream>>>(nullptr, nullptr, nullptr, nullptr, e_b1, 48, 64, BE1);
    prep_bias<<<1, 64, 0, stream>>>(nullptr, nullptr, nullptr, nullptr, e_b2, 64, 64, BE2);
    prep_bias<<<1, 64, 0, stream>>>(nullptr, nullptr, nullptr, nullptr, e_b3, 32, 32, BE3);

    // ---- stacked input (grid-sample + concat, f32 -> bf16, chunk-32 layout) ----
    {
        int total = P * HW1 * 36;
        build_stack<<<cdiv(total, 256), 256, 0, stream>>>(f1, f2, coords, X0, total);
    }

    // ---- matching net convs on WMMA ----
    const int sX0 = 9 * HW1 * 32, sY1 = 3 * HW1 * 32, sY2 = 4 * HW2 * 32;
    const int sY4 = 2 * HW2 * 32, sE  = 2 * HW1 * 32, sE3 = 1 * HW1 * 32;
    conv_gemm_wmma<<<dim3(3, H1, P), 256, 0, stream>>>(X0, sX0, H1, W1, WF1, B1, Y1, sY1, H1, W1, 9, 12, 6, 9, 1, 1, 1);
    conv_gemm_wmma<<<dim3(4, H2, P), 256, 0, stream>>>(Y1, sY1, H1, W1, WF2, B2, Y2, sY2, H2, W2, 3,  4, 8, 9, 2, 1, 1);
    conv_gemm_wmma<<<dim3(4, H2, P), 256, 0, stream>>>(Y2, sY2, H2, W2, WF3, B3, Y3, sY2, H2, W2, 4,  4, 8, 9, 1, 1, 1);
    conv_gemm_wmma<<<dim3(2, H2, P), 256, 0, stream>>>(Y3, sY2, H2, W2, WF4, B4, Y4, sY4, H2, W2, 4,  4, 4, 9, 1, 1, 1);

    // ---- embedding MLP (1x1 convs) on WMMA ----
    conv_gemm_wmma<<<dim3(2, H1, P), 256, 0, stream>>>(X0, sX0, H1, W1, WE1, BE1, E1, sE, H1, W1, 9, 12, 4, 1, 1, 0, 1);
    conv_gemm_wmma<<<dim3(2, H1, P), 256, 0, stream>>>(E1, sE, H1, W1, WE2, BE2, E2, sE, H1, W1, 2,  4, 4, 1, 1, 0, 1);
    conv_gemm_wmma<<<dim3(1, H1, P), 256, 0, stream>>>(E2, sE, H1, W1, WE3, BE3, E3, sE3, H1, W1, 2,  4, 2, 1, 1, 0, 0);

    // ---- deconv + conv6 + DAP/softmax/embedding tail ----
    deconv_block_k<<<cdiv(P * HW1 * 32, 256), 256, 0, stream>>>(Y4, w5, bn5g, bn5b, bn5m, bn5v, X5);
    conv6_cost<<<cdiv(P * HW1, 256), 256, 0, stream>>>(X5, w6, b6, costws, out);
    dap_softmax_emb<<<cdiv(HW1, 64), 64, 0, stream>>>(dap, costws, E3, out);
}
